// WordEncoder_46557445489111
// MI455X (gfx1250) — matrix operand order repaired and checked by reading, not compiled, not run on hardware
//
#include <hip/hip_runtime.h>
#include <math.h>

// ---------------------------------------------------------------------------
// Types for CDNA5 WMMA (wave32): v_wmma_f32_16x16x32_bf16
// ---------------------------------------------------------------------------
typedef __attribute__((ext_vector_type(16))) __bf16 v16bf;
typedef __attribute__((ext_vector_type(8)))  __bf16 v8bf;
typedef __attribute__((ext_vector_type(8)))  float  v8f;

#define HDIM 768
#define WP   129
#define NHEADS 4
#define HD   192
#define TT   4
#define WW   128
#define SS   512

// ---------------------------------------------------------------------------
// WMMA fragment loaders: data is K-contiguous row-major, matching the ISA
// VGPR layouts (CDNA5 ISA §7.12.2) with plain b128 loads.
// A (16x32, MxK): lane l -> row (l&15); k = kb + (l>>4)*8 + e (e<8),
//                                       k = kb + 16 + (l>>4)*8 + (e-8)
// B (32x16, KxN): lane l -> W-row (l&15); same k mapping as A
// ---------------------------------------------------------------------------
static __device__ __forceinline__ v16bf frag_a(const __bf16* __restrict__ base,
                                               int ld, int m0, int kb, int lane) {
  const __bf16* p = base + (long)(m0 + (lane & 15)) * ld + kb + ((lane >> 4) << 3);
  v8bf lo = *(const v8bf*)p;
  v8bf hi = *(const v8bf*)(p + 16);
  v16bf r;
#pragma unroll
  for (int i = 0; i < 8; ++i) { r[i] = lo[i]; r[i + 8] = hi[i]; }
  return r;
}

static __device__ __forceinline__ v16bf frag_b(const __bf16* __restrict__ base,
                                               int ld, int n0, int kb, int lane) {
  const __bf16* p = base + (long)(n0 + (lane & 15)) * ld + kb + ((lane >> 4) << 3);
  v8bf lo = *(const v8bf*)p;
  v8bf hi = *(const v8bf*)(p + 16);
  v16bf r;
#pragma unroll
  for (int i = 0; i < 8; ++i) { r[i] = lo[i]; r[i + 8] = hi[i]; }
  return r;
}

static __device__ __forceinline__ v8f wmma_bf16(v16bf a, v16bf b, v8f c) {
  return __builtin_amdgcn_wmma_f32_16x16x32_bf16(false, a, false, b, (short)0, c,
                                                 false, false);
}

// ---------------------------------------------------------------------------
// GEMM: C[M,N] = A[M,K] @ W[N,K]^T + bias, optional ReLU, f32 and/or bf16 out.
// Block: 4 waves, block tile 64(M) x 128(N), wave tile 32x64 in a 2x2 wave
// grid (waves sharing M reuse A cachelines, waves sharing N reuse B).
// Explicit double-buffered K pipeline: next K-step's 2 A + 4 B fragments are
// in flight while the 8 WMMAs of the current step execute, so waits become
// partial (loadcnt <= outstanding-next) instead of loadcnt==0 per WMMA.
// All dims divide tiles exactly -> no divergence, EXEC all-1s for WMMA.
// ---------------------------------------------------------------------------
__global__ __launch_bounds__(128)
void gemm_bf16_wmma(const __bf16* __restrict__ A, const __bf16* __restrict__ W,
                    const float* __restrict__ bias,
                    float* __restrict__ Cf, __bf16* __restrict__ Cb,
                    int M, int N, int K, int relu) {
  const int lane = threadIdx.x & 31;
  const int wave = threadIdx.x >> 5;
  const int m0 = blockIdx.y * 64 + (wave & 1) * 32;
  const int n0 = blockIdx.x * 128 + (wave >> 1) * 64;
  (void)M;

  v8f acc[2][4] = {};

  // Pipeline prologue: stage K-step 0.
  v16bf a0 = frag_a(A, K, m0 +  0, 0, lane);
  v16bf a1 = frag_a(A, K, m0 + 16, 0, lane);
  v16bf b0 = frag_b(W, K, n0 +  0, 0, lane);
  v16bf b1 = frag_b(W, K, n0 + 16, 0, lane);
  v16bf b2 = frag_b(W, K, n0 + 32, 0, lane);
  v16bf b3 = frag_b(W, K, n0 + 48, 0, lane);

  for (int kb = 0; kb < K; kb += 32) {
    // Issue next stage's loads first (redundant re-load on last iter, benign).
    const int kn = (kb + 32 < K) ? (kb + 32) : kb;
    if (kb + 64 < K)  // warm L2 one more step ahead (global_prefetch_b8)
      __builtin_prefetch(A + (long)(m0 + (lane & 15)) * K + kb + 64, 0, 1);
    v16bf na0 = frag_a(A, K, m0 +  0, kn, lane);
    v16bf na1 = frag_a(A, K, m0 + 16, kn, lane);
    v16bf nb0 = frag_b(W, K, n0 +  0, kn, lane);
    v16bf nb1 = frag_b(W, K, n0 + 16, kn, lane);
    v16bf nb2 = frag_b(W, K, n0 + 32, kn, lane);
    v16bf nb3 = frag_b(W, K, n0 + 48, kn, lane);

    // 8 WMMAs on the already-resident fragments.
    acc[0][0] = wmma_bf16(a0, b0, acc[0][0]);
    acc[1][0] = wmma_bf16(a1, b0, acc[1][0]);
    acc[0][1] = wmma_bf16(a0, b1, acc[0][1]);
    acc[1][1] = wmma_bf16(a1, b1, acc[1][1]);
    acc[0][2] = wmma_bf16(a0, b2, acc[0][2]);
    acc[1][2] = wmma_bf16(a1, b2, acc[1][2]);
    acc[0][3] = wmma_bf16(a0, b3, acc[0][3]);
    acc[1][3] = wmma_bf16(a1, b3, acc[1][3]);

    a0 = na0; a1 = na1; b0 = nb0; b1 = nb1; b2 = nb2; b3 = nb3;
  }

  // Epilogue. C/D layout: m = r + (lane>>4)*8, n = lane&15 (ISA §7.12.2)
  const int mb = m0 + ((lane >> 4) << 3);
  const int nb = n0 + (lane & 15);
#pragma unroll
  for (int mt = 0; mt < 2; ++mt) {
#pragma unroll
    for (int nt = 0; nt < 4; ++nt) {
      const int n = nb + nt * 16;
      const float bv = bias ? bias[n] : 0.0f;
#pragma unroll
      for (int r = 0; r < 8; ++r) {
        float v = acc[mt][nt][r] + bv;
        if (relu) v = v > 0.0f ? v : 0.0f;
        const long idx = (long)(mb + mt * 16 + r) * N + n;
        if (Cf) Cf[idx] = v;
        if (Cb) Cb[idx] = (__bf16)v;
      }
    }
  }
}

// ---------------------------------------------------------------------------
// f32 -> bf16 weight conversion
// ---------------------------------------------------------------------------
__global__ void f32_to_bf16(const float* __restrict__ src, __bf16* __restrict__ dst, long n) {
  long i = (long)blockIdx.x * blockDim.x + threadIdx.x;
  if (i < n) dst[i] = (__bf16)src[i];
}

// ---------------------------------------------------------------------------
// Gather + masked mean + concat cls: builds x (B*129, 768) in f32 and bf16.
// One block per output row.
// ---------------------------------------------------------------------------
__global__ __launch_bounds__(256)
void build_x(const float* __restrict__ hidden, const int* __restrict__ token_idx,
             const unsigned char* __restrict__ token_mask,
             float* __restrict__ X, __bf16* __restrict__ Xb) {
  const long row = blockIdx.x;
  const int b = (int)(row / WP);
  const int j = (int)(row % WP);
  float* xo = X + row * HDIM;
  __bf16* xbo = Xb + row * HDIM;

  if (j == 0) {  // cls = hidden[b, 0, :]
    const float* src = hidden + (long)b * SS * HDIM;
    for (int h = threadIdx.x; h < HDIM; h += blockDim.x) {
      float v = src[h];
      xo[h] = v; xbo[h] = (__bf16)v;
    }
  } else {       // masked mean over T gathered tokens
    const int w = j - 1;
    const int* idx = token_idx + ((long)b * WW + w) * TT;
    const unsigned char* mk = token_mask + ((long)b * WW + w) * TT;
    float cnt = 0.0f;
#pragma unroll
    for (int t = 0; t < TT; ++t) cnt += mk[t] ? 1.0f : 0.0f;
    const float inv = 1.0f / fmaxf(cnt, 1.0f);
    for (int h = threadIdx.x; h < HDIM; h += blockDim.x) {
      float s = 0.0f;
#pragma unroll
      for (int t = 0; t < TT; ++t)
        if (mk[t]) s += hidden[((long)b * SS + idx[t]) * HDIM + h];
      float v = s * inv;
      xo[h] = v; xbo[h] = (__bf16)v;
    }
  }
}

// ---------------------------------------------------------------------------
// Attention: one block per (query position, head, batch). 129 keys, hd=192.
// qkv row layout: [q(768) | k(768) | v(768)], head h owns cols [h*192,(h+1)*192)
// Softmax reductions are tree-based; output written as bf16 (next GEMM input).
// ---------------------------------------------------------------------------
__global__ __launch_bounds__(256)
void attention(const float* __restrict__ qkv, __bf16* __restrict__ Ob) {
  const int s = blockIdx.x, h = blockIdx.y, b = blockIdx.z;
  const int tid = threadIdx.x;
  const float* base = qkv + (long)b * WP * (3 * HDIM);
  const float* qrow = base + (long)s * (3 * HDIM) + h * HD;

  __shared__ float qs[HD];
  __shared__ float attnv[WP];
  __shared__ float red[256];

  for (int d = tid; d < HD; d += 256) qs[d] = qrow[d];
  __syncthreads();

  const float scale = rsqrtf((float)HD);
  float lmax = -INFINITY;
  for (int t = tid; t < WP; t += 256) {
    const float* krow = base + (long)t * (3 * HDIM) + HDIM + h * HD;
    float acc = 0.0f;
    for (int d = 0; d < HD; ++d) acc += qs[d] * krow[d];
    acc *= scale;
    attnv[t] = acc;
    lmax = fmaxf(lmax, acc);
  }
  red[tid] = lmax; __syncthreads();
  for (int st = 128; st > 0; st >>= 1) {
    if (tid < st) red[tid] = fmaxf(red[tid], red[tid + st]);
    __syncthreads();
  }
  const float mx = red[0];
  __syncthreads();

  float lsum = 0.0f;
  for (int t = tid; t < WP; t += 256) {
    float e = __expf(attnv[t] - mx);
    attnv[t] = e;
    lsum += e;
  }
  red[tid] = lsum; __syncthreads();
  for (int st = 128; st > 0; st >>= 1) {
    if (tid < st) red[tid] += red[tid + st];
    __syncthreads();
  }
  const float sinv = 1.0f / red[0];

  __bf16* orow = Ob + ((long)b * WP + s) * HDIM + h * HD;
  for (int d = tid; d < HD; d += 256) {
    float acc = 0.0f;
    for (int t = 0; t < WP; ++t)
      acc += attnv[t] * base[(long)t * (3 * HDIM) + 2 * HDIM + h * HD + d];
    orow[d] = (__bf16)(acc * sinv);
  }
}

// ---------------------------------------------------------------------------
// x = LayerNorm(x + add) in place; optional bf16 copy for the next GEMM.
// One block per row.
// ---------------------------------------------------------------------------
__global__ __launch_bounds__(256)
void residual_ln(float* __restrict__ X, const float* __restrict__ Add,
                 const float* __restrict__ g, const float* __restrict__ beta,
                 __bf16* __restrict__ Xb) {
  __shared__ float buf[HDIM];
  __shared__ float red[256];
  const int tid = threadIdx.x;
  const long row = blockIdx.x;
  float* x = X + row * HDIM;
  const float* a = Add + row * HDIM;

  float ls = 0.0f;
  for (int h = tid; h < HDIM; h += 256) { float v = x[h] + a[h]; buf[h] = v; ls += v; }
  red[tid] = ls; __syncthreads();
  for (int s = 128; s > 0; s >>= 1) { if (tid < s) red[tid] += red[tid + s]; __syncthreads(); }
  const float mu = red[0] * (1.0f / HDIM);
  __syncthreads();

  float lv = 0.0f;
  for (int h = tid; h < HDIM; h += 256) { float d = buf[h] - mu; lv += d * d; }
  red[tid] = lv; __syncthreads();
  for (int s = 128; s > 0; s >>= 1) { if (tid < s) red[tid] += red[tid + s]; __syncthreads(); }
  const float inv = rsqrtf(red[0] * (1.0f / HDIM) + 1e-5f);

  for (int h = tid; h < HDIM; h += 256) {
    float v = (buf[h] - mu) * inv * g[h] + beta[h];
    x[h] = v;
    if (Xb) Xb[row * HDIM + h] = (__bf16)v;
  }
}

// ---------------------------------------------------------------------------
// Head: pooled = x[b,0,:]; logits = pooled @ cls_w^T + cls_b.
// d_out = [logits (B*2) | pooled (B*768)]
// ---------------------------------------------------------------------------
__global__ __launch_bounds__(256)
void final_head(const float* __restrict__ X, const float* __restrict__ cls_w,
                const float* __restrict__ cls_b, float* __restrict__ out, int B) {
  __shared__ float r0[256], r1[256];
  const int b = blockIdx.x, tid = threadIdx.x;
  const float* row = X + (long)b * WP * HDIM;  // position 0 of batch b
  float l0 = 0.0f, l1 = 0.0f;
  for (int h = tid; h < HDIM; h += 256) {
    float v = row[h];
    out[(long)B * 2 + (long)b * HDIM + h] = v;
    l0 += v * cls_w[h];
    l1 += v * cls_w[HDIM + h];
  }
  r0[tid] = l0; r1[tid] = l1; __syncthreads();
  for (int s = 128; s > 0; s >>= 1) {
    if (tid < s) { r0[tid] += r0[tid + s]; r1[tid] += r1[tid + s]; }
    __syncthreads();
  }
  if (tid == 0) {
    out[(long)b * 2 + 0] = r0[0] + cls_b[0];
    out[(long)b * 2 + 1] = r1[0] + cls_b[1];
  }
}

// ---------------------------------------------------------------------------
// Orchestration
// ---------------------------------------------------------------------------
extern "C" void kernel_launch(void* const* d_in, const int* in_sizes, int n_in,
                              void* d_out, int out_size, void* d_ws, size_t ws_size,
                              hipStream_t stream) {
  (void)in_sizes; (void)n_in; (void)out_size; (void)ws_size;
  const float* hidden    = (const float*)d_in[0];
  const int*   token_idx = (const int*)d_in[1];
  const unsigned char* token_mask = (const unsigned char*)d_in[2];
  const float* in_proj_w = (const float*)d_in[3];
  const float* in_proj_b = (const float*)d_in[4];
  const float* out_w     = (const float*)d_in[5];
  const float* out_b     = (const float*)d_in[6];
  const float* ln1_g     = (const float*)d_in[7];
  const float* ln1_b     = (const float*)d_in[8];
  const float* ff1_w     = (const float*)d_in[9];
  const float* ff1_b     = (const float*)d_in[10];
  const float* ff2_w     = (const float*)d_in[11];
  const float* ff2_b     = (const float*)d_in[12];
  const float* ln2_g     = (const float*)d_in[13];
  const float* ln2_b     = (const float*)d_in[14];
  const float* cls_w     = (const float*)d_in[15];
  const float* cls_b     = (const float*)d_in[16];

  const int  B = 64, FF = 2048, H3 = 3 * HDIM;
  const long M = (long)B * WP;  // 8256 = 129 * 64

  // Workspace layout (256B aligned). qkv buffer is recycled as FFN hidden bf16.
  char* ws = (char*)d_ws;
  size_t off = 0;
  auto take = [&](size_t bytes) -> void* {
    void* p = ws + off;
    off += (bytes + 255) & ~(size_t)255;
    return p;
  };
  float*  x_f32    = (float*) take((size_t)M * HDIM * sizeof(float));   // x, then x1, then x2 (in place)
  __bf16* x_bf     = (__bf16*)take((size_t)M * HDIM * sizeof(__bf16));  // bf16 copies of x / x1
  float*  qkv      = (float*) take((size_t)M * H3   * sizeof(float));   // qkv; reused for h_bf16
  __bf16* o_bf     = (__bf16*)take((size_t)M * HDIM * sizeof(__bf16));  // attention output
  float*  tmp_f32  = (float*) take((size_t)M * HDIM * sizeof(float));   // attn-proj out, then ffn out
  __bf16* w_in_bf  = (__bf16*)take((size_t)H3   * HDIM * sizeof(__bf16));
  __bf16* w_out_bf = (__bf16*)take((size_t)HDIM * HDIM * sizeof(__bf16));
  __bf16* w_ff1_bf = (__bf16*)take((size_t)FF   * HDIM * sizeof(__bf16));
  __bf16* w_ff2_bf = (__bf16*)take((size_t)HDIM * FF   * sizeof(__bf16));
  __bf16* h_bf = (__bf16*)qkv;  // reuse (33.8 MB fits inside 76 MB qkv region)

  // 1) Weight conversion to bf16 (weights then live in L2: 192 MB)
  { long n = (long)H3 * HDIM;   f32_to_bf16<<<dim3((unsigned)((n + 255) / 256)), 256, 0, stream>>>(in_proj_w, w_in_bf,  n); }
  { long n = (long)HDIM * HDIM; f32_to_bf16<<<dim3((unsigned)((n + 255) / 256)), 256, 0, stream>>>(out_w,     w_out_bf, n); }
  { long n = (long)FF * HDIM;   f32_to_bf16<<<dim3((unsigned)((n + 255) / 256)), 256, 0, stream>>>(ff1_w,     w_ff1_bf, n); }
  { long n = (long)HDIM * FF;   f32_to_bf16<<<dim3((unsigned)((n + 255) / 256)), 256, 0, stream>>>(ff2_w,     w_ff2_bf, n); }

  // 2) Gather + masked mean + cls concat
  build_x<<<dim3((unsigned)M), 256, 0, stream>>>(hidden, token_idx, token_mask, x_f32, x_bf);

  // 3) QKV projection: (8256 x 2304 x 768)
  gemm_bf16_wmma<<<dim3(H3 / 128, (unsigned)(M / 64)), 128, 0, stream>>>(
      x_bf, w_in_bf, in_proj_b, qkv, (__bf16*)nullptr, (int)M, H3, HDIM, 0);

  // 4) Attention (softmax in f32, output in bf16)
  attention<<<dim3(WP, NHEADS, B), 256, 0, stream>>>(qkv, o_bf);

  // 5) Output projection: (8256 x 768 x 768)
  gemm_bf16_wmma<<<dim3(HDIM / 128, (unsigned)(M / 64)), 128, 0, stream>>>(
      o_bf, w_out_bf, out_b, tmp_f32, (__bf16*)nullptr, (int)M, HDIM, HDIM, 0);

  // 6) x1 = LN(x + attn_out); bf16 copy for FFN
  residual_ln<<<dim3((unsigned)M), 256, 0, stream>>>(x_f32, tmp_f32, ln1_g, ln1_b, x_bf);

  // 7) FFN up + ReLU: (8256 x 2048 x 768), bf16 output only
  gemm_bf16_wmma<<<dim3(FF / 128, (unsigned)(M / 64)), 128, 0, stream>>>(
      x_bf, w_ff1_bf, ff1_b, (float*)nullptr, h_bf, (int)M, FF, HDIM, 1);

  // 8) FFN down: (8256 x 768 x 2048)
  gemm_bf16_wmma<<<dim3(HDIM / 128, (unsigned)(M / 64)), 128, 0, stream>>>(
      h_bf, w_ff2_bf, ff2_b, tmp_f32, (__bf16*)nullptr, (int)M, HDIM, FF, 0);

  // 9) x2 = LN(x1 + ff)
  residual_ln<<<dim3((unsigned)M), 256, 0, stream>>>(x_f32, tmp_f32, ln2_g, ln2_b, (__bf16*)nullptr);

  // 10) pooled + logits
  final_head<<<dim3(B), 256, 0, stream>>>(x_f32, cls_w, cls_b, (float*)d_out, B);
}
